// MLPwCB_88888643158262
// MI455X (gfx1250) — matrix operand order repaired and checked by reading, not compiled, not run on hardware
//
#include <hip/hip_runtime.h>
#include <cstdint>
#include <cstddef>

// ---------------------------------------------------------------------------
// MI455X (gfx1250) MLP-with-triu-gather.
// Layer-1 GEMM (111.7 GFLOP) via v_wmma_f32_16x16x32_bf16, f32 accum.
// LDS tiles double-buffered via global_load_async_to_lds_b128 (ASYNCcnt)
// when the builtin exists; plain LDS staging fallback otherwise.
// ---------------------------------------------------------------------------

#define MDIM   128
#define FEAT   8256           // 128*129/2
#define HID    1651
#define HIDP   1664           // 13*128, padded
#define BATCH  4096

#define MT     128
#define NT     128
#define KB     32
#define LDSK   40             // padded LDS k-stride (elements): (20*r)%64 distinct for r=0..15
#define KSTEPS (FEAT / KB)    // 258

static_assert(KSTEPS * KB == FEAT, "K must tile exactly");

#if defined(__AMDGCN__) && __has_builtin(__builtin_amdgcn_global_load_async_to_lds_b128) && __has_builtin(__builtin_amdgcn_s_wait_asynccnt)
#define USE_ASYNC_LDS 1
#endif

typedef __attribute__((ext_vector_type(16))) __bf16 v16bf;
typedef __attribute__((ext_vector_type(8)))  float  v8f;
typedef __attribute__((ext_vector_type(4)))  int    v4i;   // builtin's pointee type

struct U32x8 { uint4 lo, hi; };   // 32 bytes

__device__ __forceinline__ unsigned short f32_to_bf16(float f) {
  unsigned int u = __float_as_uint(f);
  u += 0x7FFFu + ((u >> 16) & 1u);          // round-to-nearest-even
  return (unsigned short)(u >> 16);
}

// A fragment (16x32 bf16): lane m = lane&15; elems 0..7 -> K=base8+0..7,
// elems 8..15 -> K=base8+16..23, base8 = (lane&16)?8:0.  (ISA 7.12.2)
__device__ __forceinline__ v16bf load_frag_a(const unsigned short* row, int lane) {
  const int base8 = (lane & 16) ? 8 : 0;
  U32x8 u;
  u.lo = *(const uint4*)(row + base8);
  u.hi = *(const uint4*)(row + base8 + 16);
  return __builtin_bit_cast(v16bf, u);
}

// B fragment (32x16 bf16): lane n = lane&15; elems 0..7 -> K=base8+0..7,
// elems 8..15 -> K=base8+16..23, base8 = (lane&16)?8:0.   (matches A)
__device__ __forceinline__ v16bf load_frag_b(const unsigned short* row, int lane) {
  const int base16 = (lane & 16) ? 8 : 0;
  U32x8 u;
  u.lo = *(const uint4*)(row + base16);
  u.hi = *(const uint4*)(row + base16 + 16);
  return __builtin_bit_cast(v16bf, u);
}

// ---------------- pack kernels ----------------

__global__ __launch_bounds__(256) void pack_x_kernel(const float* __restrict__ sim,
                                                     unsigned short* __restrict__ Xb) {
  const int b = blockIdx.x;
  const float* S = sim + (size_t)b * (MDIM * MDIM);
  unsigned short* X = Xb + (size_t)b * FEAT;
  for (int t = threadIdx.x; t < MDIM * MDIM; t += 256) {
    const int i = t >> 7, j = t & 127;
    if (j >= i) {
      const int k = i * MDIM - (i * (i - 1)) / 2 + (j - i);  // triu_indices order
      X[k] = f32_to_bf16(S[t]);
    }
  }
}

__global__ __launch_bounds__(256) void pack_w_kernel(const float* __restrict__ W1,
                                                     unsigned short* __restrict__ Wb) {
  const int n = blockIdx.x;                  // 0..HIDP-1
  unsigned short* Wr = Wb + (size_t)n * FEAT;
  if (n < HID) {
    const float* Sr = W1 + (size_t)n * FEAT;
    for (int k = threadIdx.x; k < FEAT; k += 256) Wr[k] = f32_to_bf16(Sr[k]);
  } else {
    for (int k = threadIdx.x; k < FEAT; k += 256) Wr[k] = 0;   // zero pad rows
  }
}

// ---------------- layer-1 GEMM: H = relu(Xb @ Wb^T + b1) ----------------

__global__ __launch_bounds__(256, 2) void mlp_gemm_wmma(
    const unsigned short* __restrict__ Xb,   // [BATCH][FEAT] bf16
    const unsigned short* __restrict__ Wb,   // [HIDP][FEAT]  bf16
    const float* __restrict__ b1,
    float* __restrict__ Hp)                  // [BATCH][HIDP] relu(h)
{
  __shared__ __align__(16) unsigned short As[2][MT][LDSK];
  __shared__ __align__(16) unsigned short Bs[2][NT][LDSK];

  const int tid   = threadIdx.x;
  const int lane  = tid & 31;
  const int wave  = tid >> 5;
  const int waveM = wave & 3;        // 4 waves x 32-row slab
  const int waveN = wave >> 2;       // 2 waves x 64-col slab
  const int m0 = blockIdx.x * MT;
  const int n0 = blockIdx.y * NT;

  // 512 16B chunks per matrix per k-tile (128 rows x 4 chunks), 2 per thread
  const int ch0 = tid,       r0c = ch0 >> 2, c0c = (ch0 & 3) * 8;
  const int ch1 = tid + 256, r1c = ch1 >> 2, c1c = (ch1 & 3) * 8;

  v8f acc[2][4] = {};

#ifdef USE_ASYNC_LDS
  // Signature (from clang diagnostic): (AS1 v4i*, AS3 v4i*, imm int, imm int)
  #define ASYNC_CP16(g, l)                                            \
    __builtin_amdgcn_global_load_async_to_lds_b128(                   \
        (__attribute__((address_space(1))) v4i*)(g),                  \
        (__attribute__((address_space(3))) v4i*)(l), 0, 0)

  auto issue_tile = [&](int kb, int buf) {
    const int k0 = kb * KB;
    ASYNC_CP16(Xb + (size_t)(m0 + r0c) * FEAT + k0 + c0c, &As[buf][r0c][c0c]);
    ASYNC_CP16(Xb + (size_t)(m0 + r1c) * FEAT + k0 + c1c, &As[buf][r1c][c1c]);
    ASYNC_CP16(Wb + (size_t)(n0 + r0c) * FEAT + k0 + c0c, &Bs[buf][r0c][c0c]);
    ASYNC_CP16(Wb + (size_t)(n0 + r1c) * FEAT + k0 + c1c, &Bs[buf][r1c][c1c]);
  };
#endif

  auto compute_tile = [&](int buf) {
    const int lm = lane & 15;
    const v16bf a0 = load_frag_a(&As[buf][waveM * 32 + lm][0],      lane);
    const v16bf a1 = load_frag_a(&As[buf][waveM * 32 + 16 + lm][0], lane);
#pragma unroll
    for (int ns = 0; ns < 4; ++ns) {
      const v16bf b = load_frag_b(&Bs[buf][waveN * 64 + ns * 16 + lm][0], lane);
      acc[0][ns] = __builtin_amdgcn_wmma_f32_16x16x32_bf16(
          false, a0, false, b, (short)0, acc[0][ns], false, false);
      acc[1][ns] = __builtin_amdgcn_wmma_f32_16x16x32_bf16(
          false, a1, false, b, (short)0, acc[1][ns], false, false);
    }
  };

#ifdef USE_ASYNC_LDS
  issue_tile(0, 0);
  for (int kb = 0; kb < KSTEPS; ++kb) {
    const int buf = kb & 1;
    if (kb + 1 < KSTEPS) {
      issue_tile(kb + 1, buf ^ 1);            // prefetch into other buffer
      __builtin_amdgcn_s_wait_asynccnt(4);    // current tile's 4 ops done (FIFO)
    } else {
      __builtin_amdgcn_s_wait_asynccnt(0);
    }
    __syncthreads();                          // all waves' tile data in LDS
    compute_tile(buf);
    __syncthreads();                          // done reading before next overwrite
  }
#else
  for (int kb = 0; kb < KSTEPS; ++kb) {
    const int k0 = kb * KB;
    __syncthreads();
    *(uint4*)&As[0][r0c][c0c] = *(const uint4*)(Xb + (size_t)(m0 + r0c) * FEAT + k0 + c0c);
    *(uint4*)&As[0][r1c][c1c] = *(const uint4*)(Xb + (size_t)(m0 + r1c) * FEAT + k0 + c1c);
    *(uint4*)&Bs[0][r0c][c0c] = *(const uint4*)(Wb + (size_t)(n0 + r0c) * FEAT + k0 + c0c);
    *(uint4*)&Bs[0][r1c][c1c] = *(const uint4*)(Wb + (size_t)(n0 + r1c) * FEAT + k0 + c1c);
    __syncthreads();
    compute_tile(0);
  }
#endif

  // Epilogue: C layout — VGPR v, lanes 0-15: M=v, lanes 16-31: M=v+8; N=lane&15
  const int mB = m0 + waveM * 32 + ((lane >> 4) << 3);
  const int nB = n0 + waveN * 64 + (lane & 15);
#pragma unroll
  for (int ms = 0; ms < 2; ++ms) {
#pragma unroll
    for (int ns = 0; ns < 4; ++ns) {
      const int n = nB + ns * 16;
      if (n < HID) {
        const float bias = b1[n];
#pragma unroll
        for (int v = 0; v < 8; ++v) {
          const int m = mB + ms * 16 + v;
          Hp[(size_t)m * HIDP + n] = fmaxf(acc[ms][ns][v] + bias, 0.0f);
        }
      }
    }
  }
}

// ---------------- layer 2: out[b] = sigmoid(H[b,:]·W2 + b2) ----------------

__global__ __launch_bounds__(256) void layer2_kernel(const float* __restrict__ Hp,
                                                     const float* __restrict__ W2,
                                                     const float* __restrict__ b2,
                                                     float* __restrict__ out) {
  __shared__ float red[256];
  const int b = blockIdx.x;
  float s = 0.0f;
  for (int n = threadIdx.x; n < HID; n += 256)
    s += Hp[(size_t)b * HIDP + n] * W2[n];
  red[threadIdx.x] = s;
  __syncthreads();
  for (int off = 128; off > 0; off >>= 1) {
    if (threadIdx.x < off) red[threadIdx.x] += red[threadIdx.x + off];
    __syncthreads();
  }
  if (threadIdx.x == 0) {
    const float z = red[0] + b2[0];
    out[b] = 1.0f / (1.0f + expf(-z));
  }
}

// ---------------- host launcher ----------------

extern "C" void kernel_launch(void* const* d_in, const int* in_sizes, int n_in,
                              void* d_out, int out_size, void* d_ws, size_t ws_size,
                              hipStream_t stream) {
  const float* sim = (const float*)d_in[0];   // [4096,128,128]
  const float* W1  = (const float*)d_in[1];   // [1651,8256]
  const float* b1  = (const float*)d_in[2];   // [1651]
  const float* W2  = (const float*)d_in[3];   // [1,1651]
  const float* b2  = (const float*)d_in[4];   // [1]
  float* out = (float*)d_out;                 // [4096,1]

  const size_t XB_BYTES = (size_t)BATCH * FEAT * 2;   // 67,633,152
  const size_t WB_BYTES = (size_t)HIDP * FEAT * 2;    // 27,475,968

  char* ws = (char*)d_ws;
  unsigned short* Xb = (unsigned short*)(ws);
  unsigned short* Wb = (unsigned short*)(ws + XB_BYTES);
  float*          Hp = (float*)(ws + XB_BYTES + WB_BYTES);  // [4096][1664] f32

  pack_x_kernel<<<BATCH, 256, 0, stream>>>(sim, Xb);
  pack_w_kernel<<<HIDP, 256, 0, stream>>>(W1, Wb);

  dim3 grid(BATCH / MT, HIDP / NT);   // 32 x 13
  mlp_gemm_wmma<<<grid, 256, 0, stream>>>(Xb, Wb, b1, Hp);

  layer2_kernel<<<BATCH, 256, 0, stream>>>(Hp, W2, b2, out);
}
